// MessagePassing_53644141527539
// MI455X (gfx1250) — compile-verified, matrix operand order checked by reading, not run on hardware
//
#include <hip/hip_runtime.h>
#include <math.h>

#define NN 20000
#define INF 64
#define HH 32
#define EE 200000

typedef __attribute__((ext_vector_type(16))) _Float16 v16h;
typedef __attribute__((ext_vector_type(8)))  _Float16 v8h;
typedef __attribute__((ext_vector_type(8)))  float    v8f;

// ---------------------------------------------------------------------------
// WMMA fragment maps (gfx1250 wave32, v_wmma_f32_16x16x32_f16)
// A (16-bit, 16x32): lane l -> row l&15; per-lane halves cover
//   K = (l>>4)*8 + {0..7} then 16 + (l>>4)*8 + {0..7}.
// C/D (f32 16x16): lane -> col l&15, vgpr r -> row r + 8*(l>>4).
// ---------------------------------------------------------------------------
__device__ __forceinline__ int klocal_of(int i, int khalf) {
  return (i < 8) ? (khalf * 8 + i) : (16 + khalf * 8 + (i - 8));
}

__device__ __forceinline__ v8f wmma_f16(v16h a, v16h b, v8f c) {
  return __builtin_amdgcn_wmma_f32_16x16x32_f16(false, a, false, b,
                                                (short)0, c, false, false);
}

__device__ __forceinline__ v16h pack16(float4 a, float4 b, float4 c, float4 d) {
  v16h v;
  v[0] = (_Float16)a.x; v[1] = (_Float16)a.y; v[2]  = (_Float16)a.z; v[3]  = (_Float16)a.w;
  v[4] = (_Float16)b.x; v[5] = (_Float16)b.y; v[6]  = (_Float16)b.z; v[7]  = (_Float16)b.w;
  v[8] = (_Float16)c.x; v[9] = (_Float16)c.y; v[10] = (_Float16)c.z; v[11] = (_Float16)c.w;
  v[12] = (_Float16)d.x; v[13] = (_Float16)d.y; v[14] = (_Float16)d.z; v[15] = (_Float16)d.w;
  return v;
}

// A fragment from row-major f32 matrix
__device__ __forceinline__ v16h load_afrag_f32(const float* X, int ld, int row0, int kofs) {
  const int lane = threadIdx.x & 31;
  const int row = lane & 15, khalf = lane >> 4;
  const float* p = X + (size_t)(row0 + row) * ld + kofs + khalf * 8;
  return pack16(*(const float4*)p, *(const float4*)(p + 4),
                *(const float4*)(p + 16), *(const float4*)(p + 20));
}

// A fragment from pre-converted f16 row (row stride 32 halves): 2x b128 loads
__device__ __forceinline__ v16h load_afrag_h(const _Float16* rowbase) {
  const int khalf = (threadIdx.x & 31) >> 4;
  const _Float16* p = rowbase + khalf * 8;
  v8h r0 = *(const v8h*)p;
  v8h r1 = *(const v8h*)(p + 16);
  v16h a;
#pragma unroll
  for (int i = 0; i < 8; ++i) { a[i] = r0[i]; a[8 + i] = r1[i]; }
  return a;
}

// A fragment from LDS f32 row (for C->A transpose path)
__device__ __forceinline__ v16h load_afrag_ldsf32(const float* rowbase) {
  const int khalf = (threadIdx.x & 31) >> 4;
  const float* p = rowbase + khalf * 8;
  return pack16(*(const float4*)p, *(const float4*)(p + 4),
                *(const float4*)(p + 16), *(const float4*)(p + 20));
}

// B fragment, B[k][n] = W[k*ldn + n]
__device__ __forceinline__ v16h load_bfrag_rm(const float* W, int ldn, int n) {
  const int lane = threadIdx.x & 31, khalf = lane >> 4;
  v16h b;
#pragma unroll
  for (int i = 0; i < 16; ++i) b[i] = (_Float16)W[klocal_of(i, khalf) * ldn + n];
  return b;
}

// B fragment, B[k][n] = W[n*ldk + k]  (for x @ W^T)
__device__ __forceinline__ v16h load_bfrag_tr(const float* W, int ldk, int n) {
  const int lane = threadIdx.x & 31, khalf = lane >> 4;
  v16h b;
#pragma unroll
  for (int i = 0; i < 16; ++i) b[i] = (_Float16)W[n * ldk + klocal_of(i, khalf)];
  return b;
}

// Inverse fragment map: source-linear (kl, n) -> swizzled half index within one
// 32-lane fragment: lane = khalf*16 + (n&15), element i.
__device__ __forceinline__ void frag_pos_of(int kl, int n, int* lane, int* i) {
  int khalf, ii;
  if (kl < 16) { khalf = kl >> 3; ii = kl & 7; }
  else         { khalf = (kl - 16) >> 3; ii = 8 + ((kl - 16) & 7); }
  *lane = khalf * 16 + (n & 15);
  *i = ii;
}

// ---------------------------------------------------------------------------
__global__ void k_zero(float* __restrict__ p, int n) {
  int i = blockIdx.x * blockDim.x + threadIdx.x;
  if (i < n) p[i] = 0.0f;
}

// ---------------------------------------------------------------------------
// x_emb = relu(x @ W0 + b0); writes f32 copy and f16 copy (for WMMA gathers)
// ---------------------------------------------------------------------------
__global__ void __launch_bounds__(256) k_embed(const float* __restrict__ x,
                                               const float* __restrict__ W0,
                                               const float* __restrict__ b0,
                                               float* __restrict__ xemb,
                                               _Float16* __restrict__ xembh) {
  const int w = threadIdx.x >> 5, lane = threadIdx.x & 31;
  const int nl = lane & 15, khalf = lane >> 4;
  v16h bw[2][2];
#pragma unroll
  for (int kt = 0; kt < 2; ++kt)
#pragma unroll
    for (int nt = 0; nt < 2; ++nt)
      bw[kt][nt] = load_bfrag_rm(W0 + kt * 32 * 32, 32, nt * 16 + nl);

  const int ntiles = NN / 16;
  for (int t = blockIdx.x * 8 + w; t < ntiles; t += gridDim.x * 8) {
    const int row0 = t * 16;
    v16h a0 = load_afrag_f32(x, INF, row0, 0);
    v16h a1 = load_afrag_f32(x, INF, row0, 32);
#pragma unroll
    for (int nt = 0; nt < 2; ++nt) {
      v8f c = {};
      c = wmma_f16(a0, bw[0][nt], c);
      c = wmma_f16(a1, bw[1][nt], c);
      const int col = nt * 16 + nl;
      const float bias = b0[col];
#pragma unroll
      for (int r = 0; r < 8; ++r) {
        const int node = row0 + r + 8 * khalf;
        const float v = fmaxf(c[r] + bias, 0.0f);
        xemb[(size_t)node * HH + col] = v;
        xembh[(size_t)node * HH + col] = (_Float16)v;
      }
    }
  }
}

// ---------------------------------------------------------------------------
// Fused edge kernel (two 16-edge tiles per wave iteration):
//   hidden = relu(ea@We1+be1)
//   msg[e,o] = sum_{k,h} hidden[e,k]*xs[e,h]*We2[k,h*32+o] + sum_h xs[e,h]*be2[h*32+o]
// K=1024 WMMA GEMM, We2 f16 fragment-swizzled in LDS, atomic scatter into agg.
// Fully-unrolled K loop: A-frag(kt) = hidden[kt] (register extract) * xs-frag.
// ---------------------------------------------------------------------------
__global__ void __launch_bounds__(256) k_edges(const float* __restrict__ edge_attr,
                                               const float* __restrict__ We1,
                                               const float* __restrict__ be1,
                                               const float* __restrict__ We2,
                                               const float* __restrict__ be2,
                                               const long long* __restrict__ ei,
                                               const _Float16* __restrict__ xembh,
                                               float* __restrict__ agg) {
  __shared__ v16h ldsB[2048];        // 64KB: We2 pre-swizzled into B fragments
  __shared__ v16h ldsBias[64];       // 2KB : be2 as [32,32] B fragments
  __shared__ float ldsWe1[8 * 32];
  __shared__ float ldsBe1[32];
  __shared__ _Float16 ldsHid[8 * 32 * 32];  // per-wave: 32 edges x 32 hidden
  __shared__ int ldsDst[8 * 32];

  const int tid = threadIdx.x;

  // --- stage We2 / be2 into fragment-swizzled f16 layout ---
  // iterate in SOURCE-linear order: coalesced global_load_b32, scattered b16
  // LDS stores (one-time cost).
  {
    _Float16* bh = (_Float16*)ldsB;
    for (int s = tid; s < 32768; s += 256) {
      const int kt = s >> 10, kl = (s >> 5) & 31, n = s & 31;
      int lane, i;
      frag_pos_of(kl, n, &lane, &i);
      const int nt = n >> 4;
      bh[((kt * 2 + nt) * 32 + lane) * 16 + i] = (_Float16)We2[s];
    }
    _Float16* bb = (_Float16*)ldsBias;
    for (int s = tid; s < 1024; s += 256) {
      const int kl = s >> 5, n = s & 31;
      int lane, i;
      frag_pos_of(kl, n, &lane, &i);
      const int nt = n >> 4;
      bb[(nt * 32 + lane) * 16 + i] = (_Float16)be2[s];
    }
    if (tid < 256) ldsWe1[tid] = We1[tid];
    if (tid < 32) ldsBe1[tid] = be1[tid];
  }
  __syncthreads();

  const long long* srcI = ei;
  const long long* dstI = ei + EE;
  const int w = tid >> 5, lane = tid & 31;
  const int row = lane & 15, khalf = lane >> 4;
  const int npairs = (EE / 16) / 2;          // 6250 tile-pairs
  const int stride = gridDim.x * 8;

  for (int tp = blockIdx.x * 8 + w; tp < npairs; tp += stride) {
    const int e0A = tp * 32;
    const int e0B = e0A + 16;
    __builtin_prefetch(edge_attr + (size_t)(e0A + stride * 32) * 8, 0, 0);

    // stage dst indices: lanes 0-15 -> tile A, lanes 16-31 -> tile B
    {
      const int eidx = (lane < 16) ? (e0A + lane) : (e0B + (lane - 16));
      ldsDst[w * 32 + lane] = (int)dstI[eidx];
    }

    // hidden for both tiles: lane computes row 'row', cols [khalf*16, +16)
#pragma unroll
    for (int tb = 0; tb < 2; ++tb) {
      const int e = (tb ? e0B : e0A) + row;
      const float4 ea0 = *(const float4*)(edge_attr + (size_t)e * 8);
      const float4 ea1 = *(const float4*)(edge_attr + (size_t)e * 8 + 4);
      const float ea[8] = {ea0.x, ea0.y, ea0.z, ea0.w, ea1.x, ea1.y, ea1.z, ea1.w};
      v16h hv16;
#pragma unroll
      for (int c = 0; c < 16; ++c) {
        const int col = khalf * 16 + c;
        float acc = ldsBe1[col];
#pragma unroll
        for (int j = 0; j < 8; ++j) acc = fmaf(ea[j], ldsWe1[j * 32 + col], acc);
        hv16[c] = (_Float16)fmaxf(acc, 0.0f);
      }
      *(v16h*)&ldsHid[(w * 32 + tb * 16 + row) * 32 + khalf * 16] = hv16;
    }

    // gather xs fragments directly from f16 copy (L2-resident)
    const int nodeA = (int)srcI[e0A + row];
    const int nodeB = (int)srcI[e0B + row];
    const v16h xsA = load_afrag_h(xembh + (size_t)nodeA * HH);
    const v16h xsB = load_afrag_h(xembh + (size_t)nodeB * HH);

    asm volatile("s_wait_dscnt 0x0" ::: "memory");  // intra-wave LDS ordering

    // full hidden rows to registers (4x ds_load_b128 each)
    const _Float16* hrA = ldsHid + (w * 32 + row) * 32;
    const _Float16* hrB = ldsHid + (w * 32 + 16 + row) * 32;
    const v16h hA0 = *(const v16h*)hrA;
    const v16h hA1 = *(const v16h*)(hrA + 16);
    const v16h hB0 = *(const v16h*)hrB;
    const v16h hB1 = *(const v16h*)(hrB + 16);

    // bias term: msg += xs @ Bmat(be2)
    v8f cA0 = {}, cA1 = {}, cB0 = {}, cB1 = {};
    {
      const v16h bb0 = ldsBias[0 * 32 + lane];
      const v16h bb1 = ldsBias[1 * 32 + lane];
      cA0 = wmma_f16(xsA, bb0, cA0);
      cB0 = wmma_f16(xsB, bb0, cB0);
      cA1 = wmma_f16(xsA, bb1, cA1);
      cB1 = wmma_f16(xsB, bb1, cB1);
    }

    // main K=1024 accumulation, fully unrolled: coefficient is a constant
    // register extract -> v_pk_mul_f16 only, no per-step memory ops.
#pragma unroll
    for (int kt = 0; kt < 32; ++kt) {
      const _Float16 hvA = (kt < 16) ? hA0[kt] : hA1[kt - 16];
      const _Float16 hvB = (kt < 16) ? hB0[kt] : hB1[kt - 16];
      v16h afA, afB;
#pragma unroll
      for (int i = 0; i < 16; ++i) { afA[i] = xsA[i] * hvA; afB[i] = xsB[i] * hvB; }
      const v16h b0 = ldsB[(kt * 2 + 0) * 32 + lane];
      const v16h b1 = ldsB[(kt * 2 + 1) * 32 + lane];
      cA0 = wmma_f16(afA, b0, cA0);
      cB0 = wmma_f16(afB, b0, cB0);
      cA1 = wmma_f16(afA, b1, cA1);
      cB1 = wmma_f16(afB, b1, cB1);
    }

    // scatter-add into agg[dst] (2.56MB -> L2 atomics)
#pragma unroll
    for (int r = 0; r < 8; ++r) {
      const int m = r + 8 * khalf;
      const int dnA = ldsDst[w * 32 + m];
      const int dnB = ldsDst[w * 32 + 16 + m];
      atomicAdd(&agg[(size_t)dnA * HH + (lane & 15)], cA0[r]);
      atomicAdd(&agg[(size_t)dnA * HH + 16 + (lane & 15)], cA1[r]);
      atomicAdd(&agg[(size_t)dnB * HH + (lane & 15)], cB0[r]);
      atomicAdd(&agg[(size_t)dnB * HH + 16 + (lane & 15)], cB1[r]);
    }
  }
}

// ---------------------------------------------------------------------------
// Node kernel: x_new = agg + x_emb@root + bconv; GRU gates (r,z,n) -> out
// ---------------------------------------------------------------------------
__global__ void __launch_bounds__(256) k_nodes(const float* __restrict__ xemb,
                                               const _Float16* __restrict__ xembh,
                                               const float* __restrict__ agg,
                                               const float* __restrict__ root,
                                               const float* __restrict__ bconv,
                                               const float* __restrict__ w_ih,
                                               const float* __restrict__ w_hh,
                                               const float* __restrict__ b_ih,
                                               const float* __restrict__ b_hh,
                                               float* __restrict__ out) {
  __shared__ float ldsX[8 * 16 * 32];  // per-wave C->A transpose buffer
  const int tid = threadIdx.x, w = tid >> 5, lane = tid & 31;
  const int nl = lane & 15, khalf = lane >> 4;

  // persistent register-resident weight fragments
  v16h Broot[2], Bih[6], Bhh[6];
#pragma unroll
  for (int nt = 0; nt < 2; ++nt) Broot[nt] = load_bfrag_rm(root, 32, nt * 16 + nl);
#pragma unroll
  for (int nt = 0; nt < 6; ++nt) Bih[nt] = load_bfrag_tr(w_ih, 32, nt * 16 + nl);
#pragma unroll
  for (int nt = 0; nt < 6; ++nt) Bhh[nt] = load_bfrag_tr(w_hh, 32, nt * 16 + nl);

  float* myX = ldsX + w * 512;
  const int ntiles = NN / 16;
  for (int t = blockIdx.x * 8 + w; t < ntiles; t += gridDim.x * 8) {
    const int row0 = t * 16;
    const v16h Ah = load_afrag_h(xembh + (size_t)(row0 + (lane & 15)) * HH);

    v8f xr0 = {}, xr1 = {};
    xr0 = wmma_f16(Ah, Broot[0], xr0);
    xr1 = wmma_f16(Ah, Broot[1], xr1);

    // x_new = agg + x@root + bconv, staged to LDS for re-use as A matrix
    const float bc0 = bconv[nl], bc1 = bconv[16 + nl];
#pragma unroll
    for (int r = 0; r < 8; ++r) {
      const int m = r + 8 * khalf;
      const int node = row0 + m;
      myX[m * 32 + nl] = xr0[r] + agg[(size_t)node * HH + nl] + bc0;
      myX[m * 32 + 16 + nl] = xr1[r] + agg[(size_t)node * HH + 16 + nl] + bc1;
    }
    asm volatile("s_wait_dscnt 0x0" ::: "memory");
    const v16h Ax = load_afrag_ldsf32(myX + (lane & 15) * 32);

    v8f gi[6], gh[6];
#pragma unroll
    for (int nt = 0; nt < 6; ++nt) {
      v8f z0 = {}, z1 = {};
      gi[nt] = wmma_f16(Ax, Bih[nt], z0);
      gh[nt] = wmma_f16(Ah, Bhh[nt], z1);
    }

    // gates: column tiles 0..1 (r), 2..3 (z), 4..5 (n) align elementwise
#pragma unroll
    for (int nt = 0; nt < 2; ++nt) {
      const int col = nt * 16 + nl;
      const float br = b_ih[col] + b_hh[col];
      const float bz = b_ih[32 + col] + b_hh[32 + col];
      const float bihn = b_ih[64 + col], bhhn = b_hh[64 + col];
#pragma unroll
      for (int r = 0; r < 8; ++r) {
        const int node = row0 + r + 8 * khalf;
        const float rg = 1.0f / (1.0f + __expf(-(gi[nt][r] + gh[nt][r] + br)));
        const float zg = 1.0f / (1.0f + __expf(-(gi[nt + 2][r] + gh[nt + 2][r] + bz)));
        const float ng = tanhf(gi[nt + 4][r] + bihn + rg * (gh[nt + 4][r] + bhhn));
        const float hv = xemb[(size_t)node * HH + col];
        out[(size_t)node * HH + col] = (1.0f - zg) * ng + zg * hv;
      }
    }
  }
}

// ---------------------------------------------------------------------------
extern "C" void kernel_launch(void* const* d_in, const int* in_sizes, int n_in,
                              void* d_out, int out_size, void* d_ws, size_t ws_size,
                              hipStream_t stream) {
  (void)in_sizes; (void)n_in; (void)out_size; (void)ws_size;
  const float* x = (const float*)d_in[0];
  const float* edge_attr = (const float*)d_in[1];
  const float* W0 = (const float*)d_in[2];
  const float* b0 = (const float*)d_in[3];
  const float* We1 = (const float*)d_in[4];
  const float* be1 = (const float*)d_in[5];
  const float* We2 = (const float*)d_in[6];
  const float* be2 = (const float*)d_in[7];
  const float* root = (const float*)d_in[8];
  const float* bconv = (const float*)d_in[9];
  const float* w_ih = (const float*)d_in[10];
  const float* w_hh = (const float*)d_in[11];
  const float* b_ih = (const float*)d_in[12];
  const float* b_hh = (const float*)d_in[13];
  const long long* ei = (const long long*)d_in[14];

  float* agg = (float*)d_ws;                       // [N,H] f32 scatter accumulator
  float* xemb = agg + (size_t)NN * HH;             // [N,H] f32 embedded x / GRU h
  _Float16* xembh = (_Float16*)(xemb + (size_t)NN * HH);  // [N,H] f16 copy

  float* outp = (float*)d_out;

  k_zero<<<(NN * HH + 255) / 256, 256, 0, stream>>>(agg, NN * HH);
  k_embed<<<64, 256, 0, stream>>>(x, W0, b0, xemb, xembh);
  k_edges<<<128, 256, 0, stream>>>(edge_attr, We1, be1, We2, be2, ei, xembh, agg);
  k_nodes<<<48, 256, 0, stream>>>(xemb, xembh, agg, root, bconv, w_ih, w_hh, b_ih, b_hh, outp);
}
